// Trainer_5308579578440
// MI455X (gfx1250) — matrix operand order repaired and checked by reading, not compiled, not run on hardware
//
#include <hip/hip_runtime.h>

// ---------------------------------------------------------------------------
// GRU seq2seq on MI455X (gfx1250), persistent-kernel design.
//   B=2048 rows -> 128 workgroups x 16 rows, 4 waves (wave32) per WG.
//   Each wave owns 16 of the 64 channels; WMMA f32_16x16x32_f16 does the
//   16x192x64 per-step GEMMs; h state lives in f32 accumulator registers.
//   Weights live in LDS as f16 (96KB); biases + dense1 weights hoisted to
//   registers; accumulators zero-init (SRC2 inline 0) with bias folded into
//   the gate epilogue; dense1 for step t+1 overlapped with layer-1 WMMAs
//   (4 barriers per timestep instead of 5).
// ---------------------------------------------------------------------------

#define CCH   64          // channels
#define SEQ   512         // encoder steps
#define DEC   35          // decoder steps
#define GDIM  192         // 3*C gate rows per layer
#define TPB   128         // 4 waves of 32
#define ROWS  16          // batch rows per workgroup

typedef __attribute__((ext_vector_type(16))) _Float16 v16h;
typedef __attribute__((ext_vector_type(8)))  _Float16 v8h;
typedef __attribute__((ext_vector_type(8)))  float    v8f;

// ---- LDS layout (bytes), all chunks 16B aligned ---------------------------
#define OFF_WIH   0                 // f16 [2][192][64]  49152 B
#define OFF_WHH   49152             // f16 [2][192][64]  49152 B
#define OFF_BIH   98304             // f32 [2][192]       1536 B
#define OFF_BHH   99840             // f32 [2][192]       1536 B
#define OFF_D1W   101376            // f32 [64][2]         512 B
#define OFF_D1B   101888            // f32 [64]            256 B
#define OFF_D2W   102144            // f32 [2][64]         512 B
#define OFF_D2B   102656            // f32 [2] (pad 16)     16 B
#define OFF_EMB   102672            // f32 [64]            256 B
#define OFF_X0    102928            // f16 [16][64]       2048 B
#define OFF_H0    104976            // f16 [16][64]       2048 B
#define OFF_H1    107024            // f16 [16][64]       2048 B
#define OFF_HTOP  109072            // f32 [16][64]       4096 B
#define SMEM_BYTES 113168

// ---- WMMA operand loaders (layouts per CDNA5 ISA §7.12.2) ------
// A (16x32 f16): lanes 0-15 row M=lane: elems0-7=K k0..+7, elems8-15=K k0+16..+23
//                lanes16-31 row M=lane-16: elems0-7=K k0+8.., elems8-15=K k0+24..
__device__ __forceinline__ v16h load_a16(const _Float16* base, int k0, int lane) {
    const int m  = lane & 15;
    const int hs = (lane >> 4) << 3;           // 0 or 8
    union { v16h v; v8h h[2]; } u;
    u.h[0] = *(const v8h*)(base + m * CCH + k0 + hs);
    u.h[1] = *(const v8h*)(base + m * CCH + k0 + 16 + hs);
    return u.v;
}

// B (32x16 f16): lane holds column N = rbase+(lane&15), elems0-7=K k0+hs..+7,
// elems8-15=K k0+16+hs.. (hs=8 for lanes 16-31).  B[k][n] == W[n][k], W rows contiguous.
__device__ __forceinline__ v16h load_b16(const _Float16* w, int rbase, int k0, int lane) {
    const int row = rbase + (lane & 15);
    const int kb  = k0 + ((lane >> 4) << 3);   // 0 or 8
    const _Float16* p = w + row * CCH + kb;
    union { v16h v; v8h h[2]; } u;
    u.h[0] = *(const v8h*)(p);
    u.h[1] = *(const v8h*)(p + 16);
    return u.v;
}

__device__ __forceinline__ v8f wmma16(v16h a, v16h b, v8f c) {
    // (neg_a, A, neg_b, B, c_mod, C, reuse_a, reuse_b)
    return __builtin_amdgcn_wmma_f32_16x16x32_f16(false, a, false, b, (short)0, c,
                                                  false, false);
}

__device__ __forceinline__ float fast_sigmoid(float x) {
    return 1.0f / (1.0f + __expf(-x));
}
__device__ __forceinline__ float fast_tanh(float x) {
    float t = __expf(-2.0f * x);
    return (1.0f - t) / (1.0f + t);
}

// One GRU layer step for this wave's 16 output channels (cols [c0,c0+16)).
// Phase 1: load A operands (x tile, h tile) from LDS, barrier.
// Phase 2: overlap() (e.g. dense1 for next step), 12 WMMAs into zero-init
//          accumulators, gate math in f32 with register-resident biases,
//          update f32 h state, write f16 h tile to LDS, barrier.
template <typename F>
__device__ __forceinline__ void gru_layer(
    const _Float16* __restrict__ xbuf,   // LDS [16][64] f16 (layer input)
    _Float16*       __restrict__ hbuf,   // LDS [16][64] f16 (state, updated)
    const _Float16* __restrict__ wih,    // LDS [192][64] f16
    const _Float16* __restrict__ whh,    // LDS [192][64] f16
    int c0, int lane, v8f& hreg,
    float br, float bz, float bi, float bh,   // per-lane biases (registers)
    float* __restrict__ htop,            // optional f32 [16][64] copy-out
    F&& overlap)
{
    v16h xa0 = load_a16(xbuf, 0,  lane);
    v16h xa1 = load_a16(xbuf, 32, lane);
    v16h ha0 = load_a16(hbuf, 0,  lane);
    v16h ha1 = load_a16(hbuf, 32, lane);
    __syncthreads();                     // all reads done before anyone stores

    overlap();                           // hidden under the WMMA phase

    v8f ar = {0.f, 0.f, 0.f, 0.f, 0.f, 0.f, 0.f, 0.f};
    v8f az = {0.f, 0.f, 0.f, 0.f, 0.f, 0.f, 0.f, 0.f};
    v8f ai = {0.f, 0.f, 0.f, 0.f, 0.f, 0.f, 0.f, 0.f};
    v8f ah = {0.f, 0.f, 0.f, 0.f, 0.f, 0.f, 0.f, 0.f};

    // r gate: rows [c0, c0+16)
    ar = wmma16(xa0, load_b16(wih, c0, 0,  lane), ar);
    ar = wmma16(xa1, load_b16(wih, c0, 32, lane), ar);
    ar = wmma16(ha0, load_b16(whh, c0, 0,  lane), ar);
    ar = wmma16(ha1, load_b16(whh, c0, 32, lane), ar);
    // z gate: rows [64+c0, ...)
    az = wmma16(xa0, load_b16(wih, 64 + c0, 0,  lane), az);
    az = wmma16(xa1, load_b16(wih, 64 + c0, 32, lane), az);
    az = wmma16(ha0, load_b16(whh, 64 + c0, 0,  lane), az);
    az = wmma16(ha1, load_b16(whh, 64 + c0, 32, lane), az);
    // n gate: input part (inn) and hidden part (hn) kept separate
    ai = wmma16(xa0, load_b16(wih, 128 + c0, 0,  lane), ai);
    ai = wmma16(xa1, load_b16(wih, 128 + c0, 32, lane), ai);
    ah = wmma16(ha0, load_b16(whh, 128 + c0, 0,  lane), ah);
    ah = wmma16(ha1, load_b16(whh, 128 + c0, 32, lane), ah);

    const int nc = c0 + (lane & 15);
    const int m0 = (lane >> 4) << 3;     // C/D layout: VGPR i -> row m0+i
#pragma unroll
    for (int i = 0; i < 8; ++i) {
        float r = fast_sigmoid(ar[i] + br);
        float z = fast_sigmoid(az[i] + bz);
        float n = fast_tanh(ai[i] + bi + r * (ah[i] + bh));
        float h = (1.0f - z) * n + z * hreg[i];
        hreg[i] = h;
        hbuf[(m0 + i) * CCH + nc] = (_Float16)h;
        if (htop) htop[(m0 + i) * CCH + nc] = h;
    }
    __syncthreads();                     // h tile visible to all waves
}

__global__ void __launch_bounds__(TPB, 1) gru_seq2seq(
    const float* __restrict__ pulse,
    const float* __restrict__ d1w, const float* __restrict__ d1b,
    const float* __restrict__ d2w, const float* __restrict__ d2b,
    const float* __restrict__ Wih, const float* __restrict__ Whh,
    const float* __restrict__ bih, const float* __restrict__ bhh,
    const float* __restrict__ emb,
    float* __restrict__ out)
{
    extern __shared__ char smem[];
    _Float16* s_wih  = (_Float16*)(smem + OFF_WIH);
    _Float16* s_whh  = (_Float16*)(smem + OFF_WHH);
    float*    s_bih  = (float*)(smem + OFF_BIH);
    float*    s_bhh  = (float*)(smem + OFF_BHH);
    float*    s_d1w  = (float*)(smem + OFF_D1W);
    float*    s_d1b  = (float*)(smem + OFF_D1B);
    float*    s_d2w  = (float*)(smem + OFF_D2W);
    float*    s_d2b  = (float*)(smem + OFF_D2B);
    float*    s_emb  = (float*)(smem + OFF_EMB);
    _Float16* s_x0   = (_Float16*)(smem + OFF_X0);
    _Float16* s_h0   = (_Float16*)(smem + OFF_H0);
    _Float16* s_h1   = (_Float16*)(smem + OFF_H1);
    float*    s_htop = (float*)(smem + OFF_HTOP);

    const int tid  = threadIdx.x;
    const int lane = tid & 31;
    const int wave = tid >> 5;
    const int c0   = wave << 4;          // this wave's 16 output channels
    const int wg   = blockIdx.x;

    // ---- one-time preload: weights f32 -> f16 LDS, biases/misc f32 --------
    for (int i = tid; i < 2 * GDIM * CCH; i += TPB) {
        s_wih[i] = (_Float16)Wih[i];
        s_whh[i] = (_Float16)Whh[i];
    }
    for (int i = tid; i < 2 * GDIM; i += TPB) { s_bih[i] = bih[i]; s_bhh[i] = bhh[i]; }
    if (tid < 2 * CCH) { s_d1w[tid] = d1w[tid]; s_d2w[tid] = d2w[tid]; }
    if (tid < CCH)     { s_d1b[tid] = d1b[tid]; s_emb[tid] = emb[tid]; }
    if (tid < 2)       { s_d2b[tid] = d2b[tid]; }
    for (int i = tid; i < ROWS * CCH; i += TPB) {
        s_h0[i] = (_Float16)0.0f;
        s_h1[i] = (_Float16)0.0f;
    }
    v8f h0reg = {0.f, 0.f, 0.f, 0.f, 0.f, 0.f, 0.f, 0.f};
    v8f h1reg = {0.f, 0.f, 0.f, 0.f, 0.f, 0.f, 0.f, 0.f};
    __syncthreads();

    // ---- loop-invariant hoists: per-lane biases, per-thread dense1 weights -
    const int nc = c0 + (lane & 15);
    const float br0 = s_bih[nc]        + s_bhh[nc];
    const float bz0 = s_bih[64 + nc]   + s_bhh[64 + nc];
    const float bi0 = s_bih[128 + nc];
    const float bh0 = s_bhh[128 + nc];
    const float br1 = s_bih[GDIM + nc]       + s_bhh[GDIM + nc];
    const float bz1 = s_bih[GDIM + 64 + nc]  + s_bhh[GDIM + 64 + nc];
    const float bi1 = s_bih[GDIM + 128 + nc];
    const float bh1 = s_bhh[GDIM + 128 + nc];

    const int prow = tid >> 3;           // 0..15: batch row within tile
    const int pcb  = (tid & 7) << 3;     // 8 channels per thread
    float d1wa[8], d1wb[8], d1bb[8];
#pragma unroll
    for (int j = 0; j < 8; ++j) {
        const int c = pcb + j;
        d1wa[j] = s_d1w[2 * c];
        d1wb[j] = s_d1w[2 * c + 1];
        d1bb[j] = s_d1b[c];
    }
    auto dense1_store = [&](float2 p) {
#pragma unroll
        for (int j = 0; j < 8; ++j) {
            float v = p.x * d1wa[j] + p.y * d1wb[j] + d1bb[j];
            s_x0[prow * CCH + pcb + j] = (_Float16)v;
        }
    };
    auto nop = [] {};

    const _Float16* wih1 = s_wih + GDIM * CCH;
    const _Float16* whh1 = s_whh + GDIM * CCH;

    // ---- encoder: 512 serially dependent timesteps -------------------------
    const float* prow_ptr = pulse + (size_t)(wg * ROWS + prow) * SEQ * 2;

    dense1_store(*(const float2*)(prow_ptr));        // x for t=0
    __syncthreads();
    for (int t = 0; t < SEQ; ++t) {
        float2 pn = make_float2(0.f, 0.f);
        if (t + 1 < SEQ) pn = *(const float2*)(prow_ptr + (size_t)(t + 1) * 2);
        if (t + 8 < SEQ) __builtin_prefetch(prow_ptr + (size_t)(t + 8) * 2, 0, 0);

        gru_layer(s_x0, s_h0, s_wih, s_whh, c0, lane, h0reg,
                  br0, bz0, bi0, bh0, nullptr, nop);
        // dense1 for t+1 overlapped under layer-1's WMMA phase (s_x0 is not
        // read between layer-1's post-A barrier and its trailing barrier).
        gru_layer(s_h0, s_h1, wih1, whh1, c0, lane, h1reg,
                  br1, bz1, bi1, bh1, nullptr,
                  [&] { if (t + 1 < SEQ) dense1_store(pn); });
    }

    // ---- decoder: 35 autoregressive steps ----------------------------------
    for (int s = 0; s < DEC; ++s) {
        if (s == 0) {  // sos token = embed row, broadcast to all 16 rows
            for (int i = tid; i < ROWS * CCH; i += TPB)
                s_x0[i] = (_Float16)s_emb[i & (CCH - 1)];
            __syncthreads();
        }
        const _Float16* xb = (s == 0) ? s_x0 : s_h1;  // prev top output feeds back

        gru_layer(xb,   s_h0, s_wih, s_whh, c0, lane, h0reg,
                  br0, bz0, bi0, bh0, nullptr, nop);
        gru_layer(s_h0, s_h1, wih1, whh1, c0, lane, h1reg,
                  br1, bz1, bi1, bh1, s_htop, nop);

        // dense2: pred[b][s][o] = htop . d2w[o] + d2b[o]
        if (tid < 2 * ROWS) {
            int r = tid >> 1, o = tid & 1;
            float acc = s_d2b[o];
#pragma unroll 16
            for (int c = 0; c < CCH; ++c)
                acc += s_htop[r * CCH + c] * s_d2w[o * CCH + c];
            out[((size_t)(wg * ROWS + r) * DEC + s) * 2 + o] = acc;
        }
        // next iteration's layer-0 barriers order htop reuse
    }
}

extern "C" void kernel_launch(void* const* d_in, const int* in_sizes, int n_in,
                              void* d_out, int out_size, void* d_ws, size_t ws_size,
                              hipStream_t stream) {
    (void)n_in; (void)out_size; (void)d_ws; (void)ws_size;
    const float* pulse = (const float*)d_in[0];
    const float* d1w   = (const float*)d_in[1];
    const float* d1b   = (const float*)d_in[2];
    const float* d2w   = (const float*)d_in[3];
    const float* d2b   = (const float*)d_in[4];
    const float* Wih   = (const float*)d_in[5];
    const float* Whh   = (const float*)d_in[6];
    const float* bihp  = (const float*)d_in[7];
    const float* bhhp  = (const float*)d_in[8];
    const float* emb   = (const float*)d_in[9];
    float* out = (float*)d_out;

    const int B = in_sizes[0] / (SEQ * 2);   // 2048
    const int grid = B / ROWS;               // 128 workgroups

    (void)hipFuncSetAttribute((const void*)gru_seq2seq,
                              hipFuncAttributeMaxDynamicSharedMemorySize,
                              SMEM_BYTES);

    gru_seq2seq<<<grid, TPB, SMEM_BYTES, stream>>>(
        pulse, d1w, d1b, d2w, d2b, Wih, Whh, bihp, bhhp, emb, out);
}
